// Quantizer_53626961658225
// MI455X (gfx1250) — matrix operand order repaired and checked by reading, not compiled, not run on hardware
//
#include <hip/hip_runtime.h>
#include <hip/hip_bf16.h>

typedef __attribute__((ext_vector_type(16))) _Float16 v16h;
typedef __attribute__((ext_vector_type(8)))  float    v8f;
typedef __attribute__((ext_vector_type(4)))  float    f32x4;
typedef __attribute__((ext_vector_type(4)))  unsigned u32x4;

#define D_DIM 32
#define K_CODES 4096
#define BT 65536
#define M_TILES 4
#define WAVES_PER_BLOCK 8
#define ROWS_PER_WAVE (M_TILES * 16)                     // 64
#define ROWS_PER_BLOCK (WAVES_PER_BLOCK * ROWS_PER_WAVE) // 512
#define NUM_MAIN_BLOCKS (BT / ROWS_PER_BLOCK)            // 128
#define K_TILES (K_CODES / 16)                           // 256
#define EN_BYTES (K_CODES * D_DIM * 2)                   // 256 KB (fits 320KB WGP LDS)

// ---------------------------------------------------------------------------
// Kernel 1: normalize codebook rows -> f16 (WMMA operand precision), zero loss.
// One wave32 == one codebook row (32 consecutive elements).
// ---------------------------------------------------------------------------
__global__ void __launch_bounds__(256) vq_prep(const float* __restrict__ W,
                                               _Float16* __restrict__ enh,
                                               float* __restrict__ loss) {
  int e = blockIdx.x * 256 + threadIdx.x;      // e in [0, K*D)
  float w = W[e];
  float s = w * w;
  #pragma unroll
  for (int m = 16; m >= 1; m >>= 1) s += __shfl_xor(s, m, 32);
  float r = 1.0f / fmaxf(sqrtf(s), 1e-12f);
  enh[e] = (_Float16)(w * r);
  if (e == 0) *loss = 0.0f;                    // must re-zero every call
}

// ---------------------------------------------------------------------------
// Kernel 2: cosine-similarity argmax via v_wmma_f32_16x16x32_f16.
// Whole f16 codebook staged in LDS (256KB). Each wave owns 4 A-tiles (64 rows)
// and streams 256 B-tiles from LDS; D=32 == WMMA K so one WMMA per tile pair.
// Ping-pong pipeline (2 tiles per loop body, alternating {b0,c0}/{b1,c1}):
// no register rotation copies, compares always >=4 WMMAs behind their
// producer (no WMMA->VALU hazard NOPs), ds loads retire a stage early.
// ---------------------------------------------------------------------------
__global__ void __launch_bounds__(256) vq_argmax(const float* __restrict__ x,
                                                 const _Float16* __restrict__ enh,
                                                 float* __restrict__ idx_out) {
  extern __shared__ _Float16 en_s[];
  const int tid   = threadIdx.x;
  const int lane  = tid & 31;
  const int wave  = tid >> 5;
  const int lrow  = lane & 15;    // M (A) / N (B) within tile
  const int khalf = lane >> 4;    // which K-half this lane holds

  // Stage entire normalized codebook into LDS (global b128 -> ds b128).
  {
    const u32x4* src = (const u32x4*)enh;
    u32x4*       dst = (u32x4*)en_s;
    #pragma unroll 4
    for (int i = tid; i < EN_BYTES / 16; i += 256) dst[i] = src[i];
  }

  // Load + L2-normalize 4 A-tiles. A-layout (16-bit 16x32), contiguous K order:
  // lanes 0-15 hold K=0-15 ; lanes 16-31 hold K=16-31.
  const int base_row = (blockIdx.x * WAVES_PER_BLOCK + wave) * ROWS_PER_WAVE;
  v16h a[M_TILES];
  #pragma unroll
  for (int mt = 0; mt < M_TILES; ++mt) {
    int row = base_row + mt * 16 + lrow;
    const float* xp = x + (size_t)row * D_DIM + khalf * 16;
    f32x4 c0 = *(const f32x4*)(xp + 0);
    f32x4 c1 = *(const f32x4*)(xp + 4);
    f32x4 c2 = *(const f32x4*)(xp + 8);
    f32x4 c3 = *(const f32x4*)(xp + 12);
    float s = c0.x*c0.x + c0.y*c0.y + c0.z*c0.z + c0.w*c0.w
            + c1.x*c1.x + c1.y*c1.y + c1.z*c1.z + c1.w*c1.w
            + c2.x*c2.x + c2.y*c2.y + c2.z*c2.z + c2.w*c2.w
            + c3.x*c3.x + c3.y*c3.y + c3.z*c3.z + c3.w*c3.w;
    s += __shfl_xor(s, 16, 32);                 // partner lane holds other half-row
    float r = 1.0f / fmaxf(sqrtf(s), 1e-12f);
    a[mt][0]=(_Float16)(c0.x*r); a[mt][1]=(_Float16)(c0.y*r);
    a[mt][2]=(_Float16)(c0.z*r); a[mt][3]=(_Float16)(c0.w*r);
    a[mt][4]=(_Float16)(c1.x*r); a[mt][5]=(_Float16)(c1.y*r);
    a[mt][6]=(_Float16)(c1.z*r); a[mt][7]=(_Float16)(c1.w*r);
    a[mt][8]=(_Float16)(c2.x*r); a[mt][9]=(_Float16)(c2.y*r);
    a[mt][10]=(_Float16)(c2.z*r); a[mt][11]=(_Float16)(c2.w*r);
    a[mt][12]=(_Float16)(c3.x*r); a[mt][13]=(_Float16)(c3.y*r);
    a[mt][14]=(_Float16)(c3.z*r); a[mt][15]=(_Float16)(c3.w*r);
  }

  float bv[M_TILES][8];
  int   bi[M_TILES][8];
  #pragma unroll
  for (int mt = 0; mt < M_TILES; ++mt)
    #pragma unroll
    for (int r = 0; r < 8; ++r) { bv[mt][r] = -3.0e38f; bi[mt][r] = 0; }

  __syncthreads();

  // B operand (16-bit 32x16), same contiguous K order as A: lane n = column n, K = khalf*16 .. +15,
  // i.e. a contiguous 32-byte read of en_s row `col` -> 2x ds_load_b128.
  auto ldb = [&](int kt) -> v16h {
    return *(const v16h*)(en_s + (size_t)(kt * 16 + lrow) * D_DIM + khalf * 16);
  };
  auto upd = [&](v8f (&cc)[M_TILES], int col) {
    #pragma unroll
    for (int mt = 0; mt < M_TILES; ++mt)
      #pragma unroll
      for (int r = 0; r < 8; ++r) {    // slot r: row (r + 8*khalf), col
        float cv = cc[mt][r];
        bool gt = cv > bv[mt][r];      // strict > keeps earliest index (argmin tie rule)
        bv[mt][r] = gt ? cv : bv[mt][r];
        bi[mt][r] = gt ? col : bi[mt][r];
      }
  };

  // Prologue: tile 0 in flight in c0, b1 prefetched.
  v16h b0 = ldb(0);
  v8f c0[M_TILES], c1[M_TILES];
  #pragma unroll
  for (int mt = 0; mt < M_TILES; ++mt) {
    v8f z = {};
    c0[mt] = __builtin_amdgcn_wmma_f32_16x16x32_f16(
        false, a[mt], false, b0, (short)0, z, false, false);
  }
  v16h b1 = ldb(1);

  // Steady state: 2 tiles per iteration, ping-pong register sets.
  for (int kt = 0; kt < K_TILES - 2; kt += 2) {
    // -- even stage: compute tile kt+1 (c1 <- b1), compare tile kt (c0)
    b0 = ldb(kt + 2);
    #pragma unroll
    for (int mt = 0; mt < M_TILES; ++mt) {
      v8f z = {};
      c1[mt] = __builtin_amdgcn_wmma_f32_16x16x32_f16(
          false, a[mt], false, b1, (short)0, z, false, false);
    }
    upd(c0, kt * 16 + lrow);
    // -- odd stage: compute tile kt+2 (c0 <- b0), compare tile kt+1 (c1)
    b1 = ldb(kt + 3 < K_TILES ? kt + 3 : 0);
    #pragma unroll
    for (int mt = 0; mt < M_TILES; ++mt) {
      v8f z = {};
      c0[mt] = __builtin_amdgcn_wmma_f32_16x16x32_f16(
          false, a[mt], false, b0, (short)0, z, false, false);
    }
    upd(c1, (kt + 1) * 16 + lrow);
  }
  // Epilogue: c0 = tile K_TILES-2 in flight, b1 = tile K_TILES-1 loaded.
  #pragma unroll
  for (int mt = 0; mt < M_TILES; ++mt) {
    v8f z = {};
    c1[mt] = __builtin_amdgcn_wmma_f32_16x16x32_f16(
        false, a[mt], false, b1, (short)0, z, false, false);
  }
  upd(c0, (K_TILES - 2) * 16 + lrow);
  upd(c1, (K_TILES - 1) * 16 + lrow);

  // Reduce argmax across the 16 column-lanes (xor 8/4/2/1 stays in-group).
  #pragma unroll
  for (int mt = 0; mt < M_TILES; ++mt) {
    #pragma unroll
    for (int r = 0; r < 8; ++r) {
      float v = bv[mt][r]; int i = bi[mt][r];
      #pragma unroll
      for (int m = 8; m >= 1; m >>= 1) {
        float ov = __shfl_xor(v, m, 32);
        int   oi = __shfl_xor(i, m, 32);
        bool take = (ov > v) || (ov == v && oi < i);
        v = take ? ov : v;
        i = take ? oi : i;
      }
      bi[mt][r] = i;
    }
  }

  // lane 0 owns rows +0..7 (khalf=0), lane 16 owns rows +8..15 (khalf=1).
  if (lrow == 0) {
    #pragma unroll
    for (int mt = 0; mt < M_TILES; ++mt)
      #pragma unroll
      for (int r = 0; r < 8; ++r)
        idx_out[base_row + mt * 16 + khalf * 8 + r] = (float)bi[mt][r];
  }
}

// ---------------------------------------------------------------------------
// Kernel 3: gather W[idx] (un-normalized), recompute xn in fp32, emit
// quantized output and accumulate 1.25*mean((W[idx]-xn)^2).
// ---------------------------------------------------------------------------
__global__ void __launch_bounds__(256) vq_finalize(const float* __restrict__ x,
                                                   const float* __restrict__ W,
                                                   const float* __restrict__ idxf,
                                                   float* __restrict__ qout,
                                                   float* __restrict__ loss) {
  int e   = blockIdx.x * 256 + threadIdx.x;     // element in [0, BT*D)
  int row = e >> 5;
  int col = e & 31;
  int k   = (int)idxf[row];
  float q  = W[k * D_DIM + col];
  float xi = x[e];
  float s = xi * xi;
  #pragma unroll
  for (int m = 16; m >= 1; m >>= 1) s += __shfl_xor(s, m, 32);
  float xn = xi / fmaxf(sqrtf(s), 1e-12f);
  qout[e] = q;
  float d = q - xn;
  float p = d * d;
  #pragma unroll
  for (int m = 16; m >= 1; m >>= 1) p += __shfl_xor(p, m, 32);
  __shared__ float wsum[8];
  if ((threadIdx.x & 31) == 0) wsum[threadIdx.x >> 5] = p;
  __syncthreads();
  if (threadIdx.x == 0) {
    float t = 0.0f;
    #pragma unroll
    for (int i = 0; i < 8; ++i) t += wsum[i];
    atomicAdd(loss, t * (1.25f / (float)(BT * D_DIM)));
  }
}

// ---------------------------------------------------------------------------
extern "C" void kernel_launch(void* const* d_in, const int* in_sizes, int n_in,
                              void* d_out, int out_size, void* d_ws, size_t ws_size,
                              hipStream_t stream) {
  const float* x = (const float*)d_in[0];   // [B,T,D] fp32
  const float* W = (const float*)d_in[1];   // [K,D]  fp32
  float* out  = (float*)d_out;
  float* qout = out;                        // [B,T,D] = 2,097,152 floats
  float* idxf = out + (size_t)BT * D_DIM;   // [B,T]   = 65,536 (indices as float)
  float* loss = idxf + BT;                  // scalar
  _Float16* enh = (_Float16*)d_ws;          // 256 KB normalized f16 codebook

  vq_prep<<<(K_CODES * D_DIM) / 256, 256, 0, stream>>>(W, enh, loss);
  vq_argmax<<<NUM_MAIN_BLOCKS, 256, EN_BYTES, stream>>>(x, enh, idxf);
  vq_finalize<<<(BT * D_DIM) / 256, 256, 0, stream>>>(x, W, idxf, qout, loss);
}
